// GATEncoder_45827301048548
// MI455X (gfx1250) — compile-verified, matrix operand order checked by reading, not run on hardware
//
#include <hip/hip_runtime.h>
#include <cstdint>

#define EPS_BN 1e-5f
#define NEG_SLOPE 0.2f

typedef __attribute__((ext_vector_type(16))) __bf16 v16bf;
typedef __attribute__((ext_vector_type(8)))  __bf16 v8bf;
typedef __attribute__((ext_vector_type(8)))  float  v8f;

// ---------------- helpers ----------------
__device__ __forceinline__ unsigned ord_encode(float f) {
    unsigned u = __float_as_uint(f);
    return (u & 0x80000000u) ? ~u : (u | 0x80000000u);
}
__device__ __forceinline__ float ord_decode(unsigned u) {
    u = (u & 0x80000000u) ? (u & 0x7FFFFFFFu) : ~u;
    return __uint_as_float(u);
}

__global__ void zero_f32(float* __restrict__ p, long long cnt) {
    long long i = (long long)blockIdx.x * blockDim.x + threadIdx.x;
    if (i < cnt) p[i] = 0.0f;
}

// ---------------- LDS-tiled GEMM via WMMA (bf16 compute, f32 accumulate) ------
// Hout[n, MOUT] = X[n,128] @ W[128, MOUT].
// Block = 256 threads = 8 waves. W is staged once per block into LDS as bf16,
// transposed (WT[col][k]) with a 136-element row stride: 272 B stays 16-byte
// aligned for ds_load_b128 while rotating banks by 4 per column.
// Each wave owns one 16-row strip and sweeps all MOUT/16 column tiles,
// reusing its A fragment across 8 (or 4) WMMAs per K-step; K fully unrolled.
template<int MOUT>
__global__ __launch_bounds__(256) void gemm_wmma(const float* __restrict__ X,
                                                 const float* __restrict__ W,
                                                 float* __restrict__ Hout, int n)
{
    constexpr int NCT = MOUT / 16;            // column tiles per row strip
    __shared__ __bf16 WT[MOUT][136];          // transposed weights (only [.][0..127] used)

    // cooperative transpose-load of W into LDS (one time; W is 16-32 KB)
    for (int i = threadIdx.x; i < 128 * MOUT; i += 256) {
        int k = i / MOUT, c = i - k * MOUT;   // W row-major [128, MOUT]
        WT[c][k] = (__bf16)W[i];
    }
    __syncthreads();

    const int wave = threadIdx.x >> 5;        // 0..7
    const int lane = threadIdx.x & 31;
    const int row0 = (blockIdx.x * 8 + wave) * 16;
    const int r  = lane & 15;
    const int kh = lane >> 4;                 // K-half select (0/1)

    int rr = row0 + r; if (rr >= n) rr = n - 1;   // clamp reads; writes guarded below
    const float* xp = X + (size_t)rr * 128;

    v8f acc[NCT];
#pragma unroll
    for (int ct = 0; ct < NCT; ++ct) acc[ct] = (v8f){};

#pragma unroll
    for (int k0 = 0; k0 < 128; k0 += 32) {
        // A fragment: 16 bf16 per lane from two 8-float contiguous spans
        float4 a0 = *(const float4*)(xp + k0 + kh * 8);
        float4 a1 = *(const float4*)(xp + k0 + kh * 8 + 4);
        float4 a2 = *(const float4*)(xp + k0 + 16 + kh * 8);
        float4 a3 = *(const float4*)(xp + k0 + 16 + kh * 8 + 4);
        v16bf a;
        a[0]=(__bf16)a0.x;  a[1]=(__bf16)a0.y;  a[2]=(__bf16)a0.z;  a[3]=(__bf16)a0.w;
        a[4]=(__bf16)a1.x;  a[5]=(__bf16)a1.y;  a[6]=(__bf16)a1.z;  a[7]=(__bf16)a1.w;
        a[8]=(__bf16)a2.x;  a[9]=(__bf16)a2.y;  a[10]=(__bf16)a2.z; a[11]=(__bf16)a2.w;
        a[12]=(__bf16)a3.x; a[13]=(__bf16)a3.y; a[14]=(__bf16)a3.z; a[15]=(__bf16)a3.w;

#pragma unroll
        for (int ct = 0; ct < NCT; ++ct) {
            const int col = ct * 16 + r;
            v8bf b0 = *(const v8bf*)(&WT[col][k0 + kh * 8]);
            v8bf b1 = *(const v8bf*)(&WT[col][k0 + 16 + kh * 8]);
            union { v16bf v; struct { v8bf lo, hi; } s; } bu;
            bu.s.lo = b0; bu.s.hi = b1;
            acc[ct] = __builtin_amdgcn_wmma_f32_16x16x32_bf16(
                false, a, false, bu.v, (short)0, acc[ct], false, false);
        }
    }

    // C/D layout: acc[ct][i] -> row (i + 8*kh), col (ct*16 + r)
    const int rbase = kh * 8;
    if (row0 + 16 <= n) {
#pragma unroll
        for (int ct = 0; ct < NCT; ++ct) {
#pragma unroll
            for (int i = 0; i < 8; ++i)
                Hout[(size_t)(row0 + rbase + i) * MOUT + ct * 16 + r] = acc[ct][i];
        }
    } else {
#pragma unroll
        for (int ct = 0; ct < NCT; ++ct) {
#pragma unroll
            for (int i = 0; i < 8; ++i) {
                int orow = row0 + rbase + i;
                if (orow < n) Hout[(size_t)orow * MOUT + ct * 16 + r] = acc[ct][i];
            }
        }
    }
}

// ---------------- per-node attention dots ----------------
template<int H, int C>
__global__ void attn_dots(const float* __restrict__ Hm,
                          const float* __restrict__ att_s,
                          const float* __restrict__ att_d,
                          float* __restrict__ as_o, float* __restrict__ ad_o, int n)
{
    int idx = blockIdx.x * blockDim.x + threadIdx.x;       // idx = node*H + h
    if (idx >= n * H) return;
    int h = idx % H;
    const float* hp = Hm + (size_t)idx * C;                // (node*H+h)*C contiguous
    float s = 0.f, d = 0.f;
#pragma unroll
    for (int c = 0; c < C; ++c) {
        float v = hp[c];
        s += v * att_s[h * C + c];
        d += v * att_d[h * C + c];
    }
    as_o[idx] = s;
    ad_o[idx] = d;
}

// ---------------- edge pass 1: leaky-relu score + segment max ----------------
template<int H>
__global__ void edge_scores(const long long* __restrict__ ei, int E, int n,
                            const float* __restrict__ as, const float* __restrict__ ad,
                            float* __restrict__ ew, unsigned* __restrict__ mu)
{
    long long idx = (long long)blockIdx.x * blockDim.x + threadIdx.x;
    int ET = E + n;
    if (idx >= (long long)ET * H) return;
    int et = (int)(idx / H), h = (int)(idx - (long long)et * H);
    int s, d;
    if (et < E) { s = (int)ei[et]; d = (int)ei[(size_t)E + et]; }
    else        { s = d = et - E; }                        // self-loop
    float e = as[(size_t)s * H + h] + ad[(size_t)d * H + h];
    e = e > 0.f ? e : NEG_SLOPE * e;
    ew[idx] = e;
    atomicMax(&mu[(size_t)d * H + h], ord_encode(e));
}

// ---------------- edge pass 2: exp(e - max) + segment sum ----------------
template<int H>
__global__ void edge_expnorm(const long long* __restrict__ ei, int E, int n,
                             float* __restrict__ ew,
                             const unsigned* __restrict__ mu,
                             float* __restrict__ ssum)
{
    long long idx = (long long)blockIdx.x * blockDim.x + threadIdx.x;
    int ET = E + n;
    if (idx >= (long long)ET * H) return;
    int et = (int)(idx / H), h = (int)(idx - (long long)et * H);
    int d = (et < E) ? (int)ei[(size_t)E + et] : (et - E);
    float m = ord_decode(mu[(size_t)d * H + h]);
    float w = __expf(ew[idx] - m);
    ew[idx] = w;
    atomicAdd(&ssum[(size_t)d * H + h], w);
}

// ---------------- edge pass 3: normalize + weighted scatter-add ----------------
// One wave per edge; each lane owns CT/32 contiguous channels.
template<int H, int C>
__global__ void edge_aggregate(const long long* __restrict__ ei, int E, int n,
                               const float* __restrict__ ew,
                               const float* __restrict__ ssum,
                               const float* __restrict__ Hm,
                               float* __restrict__ out)
{
    constexpr int CT  = H * C;
    constexpr int CPL = CT / 32;
    long long t = (long long)blockIdx.x * blockDim.x + threadIdx.x;
    int et   = (int)(t >> 5);
    int lane = (int)(t & 31);
    int ET = E + n;
    if (et >= ET) return;
    int s, d;
    if (et < E) { s = (int)ei[et]; d = (int)ei[(size_t)E + et]; }
    else        { s = d = et - E; }
    int c0 = lane * CPL;
    int h  = c0 / C;                                       // CPL divides C -> one head per lane
    float alpha = ew[(size_t)et * H + h] / ssum[(size_t)d * H + h];
    const float* hp = Hm  + (size_t)s * CT + c0;
    float*       op = out + (size_t)d * CT + c0;
#pragma unroll
    for (int i = 0; i < CPL; ++i)
        atomicAdd(&op[i], alpha * hp[i]);
}

// ---------------- bias + BN (+ optional ReLU) ----------------
template<bool RELU>
__global__ void bn_finalize(const float* __restrict__ acc, const float* __restrict__ bias,
                            const float* __restrict__ g, const float* __restrict__ bt,
                            const float* __restrict__ mn, const float* __restrict__ vr,
                            float* __restrict__ out, int n, int CT)
{
    long long i = (long long)blockIdx.x * blockDim.x + threadIdx.x;
    if (i >= (long long)n * CT) return;
    int c = (int)(i % CT);
    float v = acc[i] + bias[c];
    v = g[c] * (v - mn[c]) * rsqrtf(vr[c] + EPS_BN) + bt[c];
    if (RELU) v = v > 0.f ? v : 0.f;
    out[i] = v;
}

// ---------------- host launch ----------------
static inline unsigned ublocks(long long work, int bs) {
    return (unsigned)((work + bs - 1) / bs);
}

extern "C" void kernel_launch(void* const* d_in, const int* in_sizes, int n_in,
                              void* d_out, int out_size, void* d_ws, size_t ws_size,
                              hipStream_t stream)
{
    const float*     x    = (const float*)d_in[0];
    const long long* ei   = (const long long*)d_in[1];
    const float*     W1   = (const float*)d_in[2];
    const float*     as1  = (const float*)d_in[3];
    const float*     ad1  = (const float*)d_in[4];
    const float*     b1   = (const float*)d_in[5];
    const float*     g1   = (const float*)d_in[6];
    const float*     bt1  = (const float*)d_in[7];
    const float*     mn1  = (const float*)d_in[8];
    const float*     vr1  = (const float*)d_in[9];
    const float*     W2   = (const float*)d_in[10];
    const float*     as2  = (const float*)d_in[11];
    const float*     ad2  = (const float*)d_in[12];
    const float*     b2   = (const float*)d_in[13];
    const float*     g2   = (const float*)d_in[14];
    const float*     bt2  = (const float*)d_in[15];
    const float*     mn2  = (const float*)d_in[16];
    const float*     vr2  = (const float*)d_in[17];

    const int N  = in_sizes[0] / 128;
    const int E  = in_sizes[1] / 2;
    const int ET = E + N;

    // workspace carve-up
    float*    h1   = (float*)d_ws;                         // [N,128] (reused as h2 [N,64])
    float*    x2   = h1   + (size_t)N * 128;               // [N,128] layer-1 output
    float*    accO = x2   + (size_t)N * 128;               // [N,128] aggregation accumulator
    float*    asb  = accO + (size_t)N * 128;               // [N,4]
    float*    adb  = asb  + (size_t)N * 4;                 // [N,4]
    unsigned* mu   = (unsigned*)(adb + (size_t)N * 4);     // [N,4]
    float*    ssum = (float*)(mu + (size_t)N * 4);         // [N,4]
    float*    ew   = ssum + (size_t)N * 4;                 // [ET,4]

    const int BS = 256;
    const unsigned gemmBlocks = (unsigned)((N + 127) / 128);

    // ---------------- layer 1: GAT(128 -> 32 x4 heads) ----------------
    gemm_wmma<128><<<gemmBlocks, 256, 0, stream>>>(x, W1, h1, N);
    attn_dots<4, 32><<<ublocks((long long)N * 4, BS), BS, 0, stream>>>(h1, as1, ad1, asb, adb, N);

    zero_f32<<<ublocks((long long)N * 4, BS), BS, 0, stream>>>((float*)mu, (long long)N * 4);
    zero_f32<<<ublocks((long long)N * 4, BS), BS, 0, stream>>>(ssum, (long long)N * 4);
    zero_f32<<<ublocks((long long)N * 128, BS), BS, 0, stream>>>(accO, (long long)N * 128);

    edge_scores<4><<<ublocks((long long)ET * 4, BS), BS, 0, stream>>>(ei, E, N, asb, adb, ew, mu);
    edge_expnorm<4><<<ublocks((long long)ET * 4, BS), BS, 0, stream>>>(ei, E, N, ew, mu, ssum);
    edge_aggregate<4, 32><<<ublocks((long long)ET * 32, BS), BS, 0, stream>>>(ei, E, N, ew, ssum, h1, accO);

    bn_finalize<true><<<ublocks((long long)N * 128, BS), BS, 0, stream>>>(
        accO, b1, g1, bt1, mn1, vr1, x2, N, 128);

    // ---------------- layer 2: GAT(128 -> 64, 1 head) ----------------
    gemm_wmma<64><<<gemmBlocks, 256, 0, stream>>>(x2, W2, h1, N);
    attn_dots<1, 64><<<ublocks((long long)N, BS), BS, 0, stream>>>(h1, as2, ad2, asb, adb, N);

    zero_f32<<<ublocks((long long)N, BS), BS, 0, stream>>>((float*)mu, (long long)N);
    zero_f32<<<ublocks((long long)N, BS), BS, 0, stream>>>(ssum, (long long)N);
    zero_f32<<<ublocks((long long)N * 64, BS), BS, 0, stream>>>(accO, (long long)N * 64);

    edge_scores<1><<<ublocks((long long)ET, BS), BS, 0, stream>>>(ei, E, N, asb, adb, ew, mu);
    edge_expnorm<1><<<ublocks((long long)ET, BS), BS, 0, stream>>>(ei, E, N, ew, mu, ssum);
    edge_aggregate<1, 64><<<ublocks((long long)ET * 32, BS), BS, 0, stream>>>(ei, E, N, ew, ssum, h1, accO);

    bn_finalize<false><<<ublocks((long long)N * 64, BS), BS, 0, stream>>>(
        accO, b2, g2, bt2, mn2, vr2, (float*)d_out, N, 64);
}
